// SelfAttention_22711787061564
// MI455X (gfx1250) — compile-verified, matrix operand order checked by reading, not run on hardware
//
#include <hip/hip_runtime.h>
#include <stdint.h>

typedef __bf16 bf16;
typedef __attribute__((ext_vector_type(16))) __bf16 v16bf;
typedef __attribute__((ext_vector_type(8)))  __bf16 v8bf;
typedef __attribute__((ext_vector_type(8)))  float  v8f;
typedef __attribute__((ext_vector_type(4)))  int    v4i;

// ---- bf16 <-> f32 (bit-exact, round-to-nearest-even) ----
__device__ inline bf16 f2bf(float f) {
    union { float f; unsigned u; } c; c.f = f;
    unsigned r = (c.u + 0x7FFFu + ((c.u >> 16) & 1u)) >> 16;
    union { unsigned short s; bf16 b; } o; o.s = (unsigned short)r;
    return o.b;
}
__device__ inline float bf2f(bf16 b) {
    union { unsigned short s; bf16 b; } i; i.b = b;
    union { unsigned u; float f; } o; o.u = ((unsigned)i.s) << 16;
    return o.f;
}

// ---- WMMA fragment load: 16x32 bf16 tile from row-major storage ----
// lane L holds row r=L&15; K-half kh=L>>4:
//   elems 0..7 = K[kh*8..kh*8+7], elems 8..15 = K[16+kh*8..16+kh*8+7]
// Also serves B-fragments when B is stored TRANSPOSED (row = N column).
__device__ inline v16bf load_frag_rm(const bf16* p, int stride) {
    int lane = threadIdx.x & 31;
    int r    = lane & 15;
    int kh   = (lane >> 4) << 3;
    const bf16* q = p + r * stride + kh;
    union { v16bf v; v8bf h[2]; } u;
    u.h[0] = *(const v8bf*)(q);
    u.h[1] = *(const v8bf*)(q + 16);
    return u.v;
}

__device__ inline v8f wmma_bf(v16bf a, v16bf b, v8f c) {
    return __builtin_amdgcn_wmma_f32_16x16x32_bf16(
        false, a, false, b, (short)0, c, false, false);
}

// ---- CDNA5 async global->LDS copy (guarded; falls back to sync copy) ----
#if defined(__has_builtin)
#if __has_builtin(__builtin_amdgcn_global_load_async_to_lds_b128) && \
    __has_builtin(__builtin_amdgcn_s_wait_asynccnt)
#define USE_ASYNC_LDS 1
#endif
#endif

#ifdef USE_ASYNC_LDS
typedef __attribute__((address_space(1))) v4i av4i_g;   // global (AS1)
typedef __attribute__((address_space(3))) v4i av4i_l;   // LDS (AS3)
__device__ inline void async_copy16(bf16* lds_dst, const bf16* gsrc) {
    __builtin_amdgcn_global_load_async_to_lds_b128(
        (av4i_g*)gsrc, (av4i_l*)lds_dst, 0, 0);
}
__device__ inline void async_wait() { __builtin_amdgcn_s_wait_asynccnt(0); }
#else
__device__ inline void async_copy16(bf16* lds_dst, const bf16* gsrc) {
    *(v8bf*)lds_dst = *(const v8bf*)gsrc;
}
__device__ inline void async_wait() {}
#endif

// ---------------- conversion kernels ----------------
__global__ void k_cvt(const float* __restrict__ src, bf16* __restrict__ dst, int n) {
    int i = blockIdx.x * 256 + threadIdx.x;
    if (i < n) dst[i] = f2bf(src[i]);
}
__global__ void k_cvt_t(const float* __restrict__ src, bf16* __restrict__ dst, int K, int N) {
    int i = blockIdx.x * 256 + threadIdx.x;
    if (i < K * N) {
        int k = i / N, n = i % N;
        dst[n * K + k] = f2bf(src[i]);
    }
}

// -------- projection GEMM: proj[32768,384] = xb @ Wcat, reg-double-buffered --
__global__ void k_proj(const bf16* __restrict__ xb, const bf16* __restrict__ wcatT,
                       bf16* __restrict__ proj) {
    int mblk  = blockIdx.x * 32;              // 1024 blocks
    int wave  = threadIdx.x >> 5;
    int mband = (wave & 1) * 16;
    int nq    = (wave >> 1) * 96;             // 4 N-quarters of 96
    const bf16* Ab = xb + (size_t)(mblk + mband) * 512;
    v8f acc[6] = {};
    v16bf a_c = load_frag_rm(Ab, 512);
    v16bf b_c[6];
#pragma unroll
    for (int t = 0; t < 6; ++t)
        b_c[t] = load_frag_rm(wcatT + (size_t)(nq + t * 16) * 512, 512);
    for (int kk = 0; kk < 512; kk += 32) {
        int kn = (kk + 32 < 512) ? kk + 32 : 0;     // wrap (redundant last prefetch)
        v16bf a_n = load_frag_rm(Ab + kn, 512);
        v16bf b_n[6];
#pragma unroll
        for (int t = 0; t < 6; ++t)
            b_n[t] = load_frag_rm(wcatT + (size_t)(nq + t * 16) * 512 + kn, 512);
#pragma unroll
        for (int t = 0; t < 6; ++t)
            acc[t] = wmma_bf(a_c, b_c[t], acc[t]);
        a_c = a_n;
#pragma unroll
        for (int t = 0; t < 6; ++t) b_c[t] = b_n[t];
    }
    int lane = threadIdx.x & 31;
    int nn = lane & 15, mb2 = (lane >> 4) << 3;
#pragma unroll
    for (int t = 0; t < 6; ++t)
#pragma unroll
        for (int j = 0; j < 8; ++j)
            proj[(size_t)(mblk + mband + mb2 + j) * 384 + nq + t * 16 + nn] = f2bf(acc[t][j]);
}

// ---------------- split theta / pool phi,g ----------------
__global__ void k_theta(const bf16* __restrict__ proj, bf16* __restrict__ theta) {
    int i = blockIdx.x * 256 + threadIdx.x;
    if (i < 32768 * 64) {
        int m = i >> 6, d = i & 63;
        theta[i] = proj[(size_t)m * 384 + d];
    }
}
__global__ void k_pool(const bf16* __restrict__ proj, bf16* __restrict__ phi,
                       bf16* __restrict__ gT) {
    int i = blockIdx.x * 256 + threadIdx.x;
    if (i >= 8 * 1024 * 320) return;
    int d  = i % 320;
    int t  = i / 320;
    int pp = t & 1023;
    int b  = t >> 10;
    int ph = pp >> 5, pw = pp & 31;
    int col = 64 + d;
    float m = -3.0e38f;
#pragma unroll
    for (int dy = 0; dy < 2; ++dy)
#pragma unroll
        for (int dx = 0; dx < 2; ++dx) {
            int p = (2 * ph + dy) * 64 + (2 * pw + dx);
            m = fmaxf(m, bf2f(proj[(size_t)(b * 4096 + p) * 384 + col]));
        }
    if (d < 64) phi[(size_t)(b * 1024 + pp) * 64 + d] = f2bf(m);
    else        gT[(size_t)b * 256 * 1024 + (size_t)(d - 64) * 1024 + pp] = f2bf(m);
}

// ------------- fused scores + softmax -> beta (bf16), B prefetched ----------
__global__ void k_attn(const bf16* __restrict__ theta, const bf16* __restrict__ phi,
                       bf16* __restrict__ beta) {
    __shared__ float S[16 * 1024];             // 64 KB
    int b    = blockIdx.x >> 8;
    int rblk = (blockIdx.x & 255) * 16;
    int wave = threadIdx.x >> 5;
    const bf16* Ab = theta + (size_t)(b * 4096 + rblk) * 64;
    const bf16* Pb = phi + (size_t)(b * 1024 + wave * 128) * 64;
    v16bf a0 = load_frag_rm(Ab +  0, 64);
    v16bf a1 = load_frag_rm(Ab + 32, 64);
    int lane = threadIdx.x & 31;
    int nn = lane & 15, mb2 = (lane >> 4) << 3;
    v16bf b0c = load_frag_rm(Pb +  0, 64);
    v16bf b1c = load_frag_rm(Pb + 32, 64);
#pragma unroll
    for (int t = 0; t < 8; ++t) {
        int tn = (t + 1) & 7;                  // wrap: redundant last prefetch
        v16bf b0n = load_frag_rm(Pb + (size_t)tn * 16 * 64 +  0, 64);
        v16bf b1n = load_frag_rm(Pb + (size_t)tn * 16 * 64 + 32, 64);
        v8f acc = {};
        acc = wmma_bf(a0, b0c, acc);
        acc = wmma_bf(a1, b1c, acc);
        int n0 = wave * 128 + t * 16;
#pragma unroll
        for (int j = 0; j < 8; ++j)
            S[(mb2 + j) * 1024 + n0 + nn] = acc[j];
        b0c = b0n; b1c = b1n;
    }
    __syncthreads();
    int row = threadIdx.x >> 4, l16 = threadIdx.x & 15;
    float mx = -3.0e38f;
    for (int c = l16; c < 1024; c += 16) mx = fmaxf(mx, S[row * 1024 + c]);
#pragma unroll
    for (int o = 8; o; o >>= 1) mx = fmaxf(mx, __shfl_xor(mx, o, 16));
    float sum = 0.0f;
    for (int c = l16; c < 1024; c += 16) {
        float e = __expf(S[row * 1024 + c] - mx);
        S[row * 1024 + c] = e;
        sum += e;
    }
#pragma unroll
    for (int o = 8; o; o >>= 1) sum += __shfl_xor(sum, o, 16);
    float inv = 1.0f / sum;
    bf16* outp = beta + (size_t)(b * 4096 + rblk + row) * 1024;
    for (int c = l16; c < 1024; c += 16) outp[c] = f2bf(S[row * 1024 + c] * inv);
}

// -------- PV GEMM: O = beta @ g, B staged in LDS (async, double-buffered) ---
// M_BLK=128 (8 M-band waves), each wave does all 16 N-tiles. OT stored d-major.
#define OG_PAD 40   // 32 bf16 payload + 8 pad: 80B rows -> conflict-free frags
__global__ void k_og(const bf16* __restrict__ beta, const bf16* __restrict__ gT,
                     bf16* __restrict__ OT) {
    __shared__ bf16 Bb[2][256 * OG_PAD];       // 2 x 20 KB
    int b    = blockIdx.x >> 5;                // 32 blocks per batch
    int nblk = (blockIdx.x & 31) * 128;
    int wave = threadIdx.x >> 5;
    int mband = wave * 16;
    const bf16* Ab = beta + (size_t)(b * 4096 + nblk + mband) * 1024;
    const bf16* Gb = gT + (size_t)b * 256 * 1024;

    v8f acc[16] = {};
    v16bf a_c = load_frag_rm(Ab, 1024);

    // coalesced cooperative fill: 1024 x 16B chunks, 4 per thread
    {
#pragma unroll
        for (int j = 0; j < 4; ++j) {
            int idx = threadIdx.x + 256 * j;
            int d = idx >> 2, c = idx & 3;
            async_copy16(&Bb[0][d * OG_PAD + c * 8], Gb + (size_t)d * 1024 + c * 8);
        }
    }
    int cur = 0;
    for (int kk = 0; kk < 1024; kk += 32) {
        async_wait();
        __syncthreads();                       // fills visible; prev reads done
        int kn = (kk + 32 < 1024) ? kk + 32 : 0;
        if (kk + 32 < 1024) {
#pragma unroll
            for (int j = 0; j < 4; ++j) {
                int idx = threadIdx.x + 256 * j;
                int d = idx >> 2, c = idx & 3;
                async_copy16(&Bb[cur ^ 1][d * OG_PAD + c * 8],
                             Gb + (size_t)d * 1024 + kn + c * 8);
            }
        }
        v16bf a_n = load_frag_rm(Ab + kn, 1024);
#pragma unroll
        for (int t = 0; t < 16; ++t) {
            v16bf bfr = load_frag_rm(&Bb[cur][t * 16 * OG_PAD], OG_PAD);
            acc[t] = wmma_bf(a_c, bfr, acc[t]);
        }
        a_c = a_n;
        cur ^= 1;
    }
    int lane = threadIdx.x & 31;
    int nn = lane & 15, mb2 = (lane >> 4) << 3;
#pragma unroll
    for (int t = 0; t < 16; ++t) {
        int d = t * 16 + nn;
        int n = nblk + mband + mb2;
        v8bf pack;
#pragma unroll
        for (int j = 0; j < 8; ++j) pack[j] = f2bf(acc[t][j]);
        *(v8bf*)(OT + (size_t)b * 256 * 4096 + (size_t)d * 4096 + n) = pack;
    }
}

// -------- output GEMM + epilogue, reg-double-buffered -----------------------
__global__ void k_out(const bf16* __restrict__ OT, const bf16* __restrict__ woT,
                      const float* __restrict__ x, const float* __restrict__ gamma,
                      float* __restrict__ out) {
    int b = blockIdx.x >> 8;
    int q = blockIdx.x & 255;
    int wave = threadIdx.x >> 5;
    const bf16* Ab = OT + (size_t)(b * 256 + q) * 4096;   // 16x256 row-major
    const bf16* Wb = woT + (size_t)wave * 64 * 256;
    v8f acc[4] = {};
    v16bf a_c = load_frag_rm(Ab, 256);
    v16bf b_c[4];
#pragma unroll
    for (int t = 0; t < 4; ++t)
        b_c[t] = load_frag_rm(Wb + (size_t)t * 16 * 256, 256);
    for (int kk = 0; kk < 256; kk += 32) {
        int kn = (kk + 32 < 256) ? kk + 32 : 0;
        v16bf a_n = load_frag_rm(Ab + kn, 256);
        v16bf b_n[4];
#pragma unroll
        for (int t = 0; t < 4; ++t)
            b_n[t] = load_frag_rm(Wb + (size_t)t * 16 * 256 + kn, 256);
#pragma unroll
        for (int t = 0; t < 4; ++t)
            acc[t] = wmma_bf(a_c, b_c[t], acc[t]);
        a_c = a_n;
#pragma unroll
        for (int t = 0; t < 4; ++t) b_c[t] = b_n[t];
    }
    float gm = gamma[0];
    int lane = threadIdx.x & 31;
    int nn = lane & 15, mb2 = (lane >> 4) << 3;
#pragma unroll
    for (int t = 0; t < 4; ++t) {
        int c = wave * 64 + t * 16 + nn;
#pragma unroll
        for (int j = 0; j < 8; ++j) {
            int p = q * 16 + mb2 + j;
            size_t idx = ((size_t)(b * 4096 + p)) * 512 + c;
            out[idx] = gm * acc[t][j] + x[idx];
        }
    }
}

extern "C" void kernel_launch(void* const* d_in, const int* in_sizes, int n_in,
                              void* d_out, int out_size, void* d_ws, size_t ws_size,
                              hipStream_t stream) {
    const float* x  = (const float*)d_in[0];
    const float* wt = (const float*)d_in[1];
    const float* wp = (const float*)d_in[2];
    const float* wg = (const float*)d_in[3];
    const float* wo = (const float*)d_in[4];
    const float* gm = (const float*)d_in[5];
    float* out = (float*)d_out;

    char* ws = (char*)d_ws;
    size_t off = 0;
    auto take = [&](size_t bytes) -> char* {
        char* p = ws + off;
        off += (bytes + 255) & ~(size_t)255;
        return p;
    };
    bf16* xb    = (bf16*)take((size_t)32768 * 512 * 2);
    bf16* wcatT = (bf16*)take((size_t)384 * 512 * 2);
    bf16* woT   = (bf16*)take((size_t)512 * 256 * 2);
    bf16* proj  = (bf16*)take((size_t)32768 * 384 * 2);
    bf16* theta = (bf16*)take((size_t)32768 * 64 * 2);
    bf16* phi   = (bf16*)take((size_t)8 * 1024 * 64 * 2);
    bf16* gT    = (bf16*)take((size_t)8 * 256 * 1024 * 2);
    bf16* beta  = (bf16*)take((size_t)8 * 4096 * 1024 * 2);
    bf16* OT    = (bf16*)take((size_t)8 * 256 * 4096 * 2);

    k_cvt  <<<(32768 * 512 + 255) / 256, 256, 0, stream>>>(x, xb, 32768 * 512);
    k_cvt_t<<<(512 * 64  + 255) / 256, 256, 0, stream>>>(wt, wcatT,             512, 64);
    k_cvt_t<<<(512 * 64  + 255) / 256, 256, 0, stream>>>(wp, wcatT + 64 * 512,  512, 64);
    k_cvt_t<<<(512 * 256 + 255) / 256, 256, 0, stream>>>(wg, wcatT + 128 * 512, 512, 256);
    k_cvt_t<<<(256 * 512 + 255) / 256, 256, 0, stream>>>(wo, woT,               256, 512);

    k_proj <<<1024, 256, 0, stream>>>(xb, wcatT, proj);
    k_theta<<<(32768 * 64 + 255) / 256, 256, 0, stream>>>(proj, theta);
    k_pool <<<(8 * 1024 * 320 + 255) / 256, 256, 0, stream>>>(proj, phi, gT);
    k_attn <<<8 * 256, 256, 0, stream>>>(theta, phi, beta);
    k_og   <<<8 * 32,  256, 0, stream>>>(beta, gT, OT);
    k_out  <<<8 * 256, 256, 0, stream>>>(OT, woT, x, gm, out);
}
